// MidBlock_661424963863
// MI455X (gfx1250) — matrix operand order repaired and checked by reading, not compiled, not run on hardware
//
#include <hip/hip_runtime.h>
#include <hip/hip_bf16.h>

typedef _Float16 h16 __attribute__((ext_vector_type(16)));
typedef _Float16 h8  __attribute__((ext_vector_type(8)));
typedef float    f8  __attribute__((ext_vector_type(8)));

#define CCH 512
#define HW  4096
#define NB  4

// ---------------- WMMA fragment loaders (CDNA5 16x16x32 f16 layouts) ----------------
// A (16x32, MxK): lane m = lane&15; halves e<8 -> k=(lane&16?8:0)+e ; e>=8 -> k=16+(lane&16?8:0)+(e&7)
__device__ __forceinline__ h16 frag_a(const _Float16* __restrict__ row, int lane) {
    const int ko = (lane & 16) ? 8 : 0;
    h8 lo = *(const h8*)(row + ko);
    h8 hi = *(const h8*)(row + ko + 16);
    h16 a;
#pragma unroll
    for (int i = 0; i < 8; ++i) { a[i] = lo[i]; a[i + 8] = hi[i]; }
    return a;
}
// B (32x16, KxN): lane n = lane&15; e<8 -> k=(lane&16?8:0)+e ; e>=8 -> k=16+(lane&16?8:0)+(e&7)
__device__ __forceinline__ h16 frag_b(const _Float16* __restrict__ row, int lane) {
    const int ko = (lane & 16) ? 8 : 0;
    h8 lo = *(const h8*)(row + ko);
    h8 hi = *(const h8*)(row + ko + 16);
    h16 b;
#pragma unroll
    for (int i = 0; i < 8; ++i) { b[i] = lo[i]; b[i + 8] = hi[i]; }
    return b;
}

// ============= Generic GEMM: C[m][n] = scale*sum_k A[m][k]*B[n][k] (+bias)(+resid) =========
// Block tile 128x128, 8 waves, each wave 32(M) x 64(N) = 8 WMMA accumulators.
// A rows: A + z*aBatch + m*K ; B rows: B + z*bBatch + n*K (f16, K-major, K%32==0)
// out: z*oBatch + m*ldOut + n (f16 or f32). biasMode: 0 none, 1 per-m, 2 per-n.
template <bool OUT_F16>
__global__ __launch_bounds__(256) void gemm_wmma(
    const _Float16* __restrict__ A, const _Float16* __restrict__ B,
    long long aBatch, long long bBatch, int K,
    void* __restrict__ outp, long long oBatch, int ldOut,
    const float* __restrict__ bias, int biasMode,
    const float* __restrict__ resid, float scale)
{
    __shared__ _Float16 Ls[2][256][40];            // rows 0..127 = A, 128..255 = B
    const int tid = threadIdx.x, lane = tid & 31, wid = tid >> 5;
    const int z = blockIdx.z;
    const long long mBase = (long long)blockIdx.x * 128, nBase = (long long)blockIdx.y * 128;
    const _Float16* Ab = A + (long long)z * aBatch;
    const _Float16* Bb = B + (long long)z * bBatch;

    // load role: threads 0..127 fill A half, 128..255 fill B half (wave-uniform)
    const int lrow = tid & 127;
    const _Float16* gbase = (tid < 128) ? (Ab + (mBase + lrow) * (long long)K)
                                        : (Bb + (nBase + lrow) * (long long)K);
    const int nK = K >> 5;

    {   // preload tile 0
        h8 r0 = *(const h8*)(gbase + 0),  r1 = *(const h8*)(gbase + 8);
        h8 r2 = *(const h8*)(gbase + 16), r3 = *(const h8*)(gbase + 24);
        _Float16* d = &Ls[0][tid][0];
        *(h8*)(d) = r0; *(h8*)(d + 8) = r1; *(h8*)(d + 16) = r2; *(h8*)(d + 24) = r3;
    }
    __syncthreads();

    const int mSub = (wid & 3) * 32;               // wave M offset (2 frag rows)
    const int nSub = (wid >> 2) * 64;              // wave N offset (4 frag rows)
    const int arow = mSub + (lane & 15);
    const int brow = 128 + nSub + (lane & 15);
    f8 acc[2][4];
#pragma unroll
    for (int i = 0; i < 2; ++i)
#pragma unroll
        for (int j = 0; j < 4; ++j) acc[i][j] = {};

    for (int s = 0; s < nK; ++s) {
        const int buf = s & 1;
        const bool more = (s + 1) < nK;
        h8 r0, r1, r2, r3;
        if (more) {                                  // issue next-tile loads early
            const _Float16* g = gbase + (long long)(s + 1) * 32;
            r0 = *(const h8*)(g + 0);  r1 = *(const h8*)(g + 8);
            r2 = *(const h8*)(g + 16); r3 = *(const h8*)(g + 24);
        }
        h16 a0 = frag_a(&Ls[buf][arow][0], lane);
        h16 a1 = frag_a(&Ls[buf][arow + 16][0], lane);
#pragma unroll
        for (int j = 0; j < 4; ++j) {
            h16 bj = frag_b(&Ls[buf][brow + j * 16][0], lane);
            acc[0][j] = __builtin_amdgcn_wmma_f32_16x16x32_f16(false, a0, false, bj, (short)0, acc[0][j], false, false);
            acc[1][j] = __builtin_amdgcn_wmma_f32_16x16x32_f16(false, a1, false, bj, (short)0, acc[1][j], false, false);
        }
        if (more) {                                  // LDS fill lands behind the WMMA pack
            _Float16* d = &Ls[buf ^ 1][tid][0];
            *(h8*)(d) = r0; *(h8*)(d + 8) = r1; *(h8*)(d + 16) = r2; *(h8*)(d + 24) = r3;
        }
        __syncthreads();
    }

    const int mOff = (lane & 16) ? 8 : 0, nCol = lane & 15;
#pragma unroll
    for (int i = 0; i < 2; ++i) {
#pragma unroll
        for (int j = 0; j < 4; ++j) {
            const long long n = nBase + nSub + j * 16 + nCol;
#pragma unroll
            for (int r = 0; r < 8; ++r) {
                const long long m = mBase + mSub + i * 16 + mOff + r;
                float v = acc[i][j][r] * scale;
                if (biasMode == 1) v += bias[m];
                else if (biasMode == 2) v += bias[n];
                const long long idx = (long long)z * oBatch + m * ldOut + n;
                if constexpr (OUT_F16) {
                    ((_Float16*)outp)[idx] = (_Float16)v;
                } else {
                    if (resid) v += resid[idx];
                    ((float*)outp)[idx] = v;
                }
            }
        }
    }
}

// ============= conv3x3 (SAME) implicit GEMM: K = 9 taps x 512 ic ============================
// act: NHWC f16 [B][64][64][512]; wpk: [oc][tap*512+ic] f16; out: NCHW f32 (+bias, +resid)
// Block tile: M=128 oc x N=128 pixels (two image rows of one batch).
__global__ __launch_bounds__(256) void conv3x3_wmma(
    const _Float16* __restrict__ act, const _Float16* __restrict__ wpk,
    const float* __restrict__ bias, const float* __restrict__ resid,
    float* __restrict__ outp)
{
    __shared__ _Float16 Ls[2][256][40];
    const int tid = threadIdx.x, lane = tid & 31, wid = tid >> 5;
    const int mBase = blockIdx.x * 128;            // oc
    const int tIdx = blockIdx.y;                   // (b, ypair)
    const int b = tIdx >> 5, yBase = (tIdx & 31) * 2;
    const int lrow = tid & 127;

    const _Float16* aRow = wpk + (long long)(mBase + lrow) * 4608;   // threads <128
    const int py = yBase + (lrow >> 6), px = lrow & 63;              // threads >=128

    auto loadTile = [&](h8& r0, h8& r1, h8& r2, h8& r3, int s) {
        if (tid < 128) {
            const _Float16* g = aRow + s * 32;
            r0 = *(const h8*)(g + 0);  r1 = *(const h8*)(g + 8);
            r2 = *(const h8*)(g + 16); r3 = *(const h8*)(g + 24);
        } else {
            // tap = s>>4 is wave-uniform; arithmetic decomposition stays in SALU
            const int tap = s >> 4, icb = (s & 15) * 32;
            const int ty = tap / 3;                // 0..2
            const int tx = tap - ty * 3;           // 0..2
            const int yy = py + ty - 1, xx = px + tx - 1;
            h8 zz = {};
            if ((unsigned)yy < 64u && (unsigned)xx < 64u) {
                const _Float16* g = act + (((long long)b * 64 + yy) * 64 + xx) * 512 + icb;
                r0 = *(const h8*)(g + 0);  r1 = *(const h8*)(g + 8);
                r2 = *(const h8*)(g + 16); r3 = *(const h8*)(g + 24);
            } else { r0 = zz; r1 = zz; r2 = zz; r3 = zz; }
        }
    };
    auto storeTile = [&](h8 r0, h8 r1, h8 r2, h8 r3, int buf) {
        _Float16* d = &Ls[buf][tid][0];
        *(h8*)(d) = r0; *(h8*)(d + 8) = r1; *(h8*)(d + 16) = r2; *(h8*)(d + 24) = r3;
    };

    { h8 r0, r1, r2, r3; loadTile(r0, r1, r2, r3, 0); storeTile(r0, r1, r2, r3, 0); }
    __syncthreads();

    const int mSub = (wid & 3) * 32, nSub = (wid >> 2) * 64;
    const int arow = mSub + (lane & 15);
    const int brow = 128 + nSub + (lane & 15);
    f8 acc[2][4];
#pragma unroll
    for (int i = 0; i < 2; ++i)
#pragma unroll
        for (int j = 0; j < 4; ++j) acc[i][j] = {};

    for (int s = 0; s < 144; ++s) {
        const int buf = s & 1;
        const bool more = (s + 1) < 144;
        h8 r0, r1, r2, r3;
        if (more) loadTile(r0, r1, r2, r3, s + 1);
        h16 a0 = frag_a(&Ls[buf][arow][0], lane);
        h16 a1 = frag_a(&Ls[buf][arow + 16][0], lane);
#pragma unroll
        for (int j = 0; j < 4; ++j) {
            h16 bj = frag_b(&Ls[buf][brow + j * 16][0], lane);
            acc[0][j] = __builtin_amdgcn_wmma_f32_16x16x32_f16(false, a0, false, bj, (short)0, acc[0][j], false, false);
            acc[1][j] = __builtin_amdgcn_wmma_f32_16x16x32_f16(false, a1, false, bj, (short)0, acc[1][j], false, false);
        }
        if (more) storeTile(r0, r1, r2, r3, buf ^ 1);
        __syncthreads();
    }

    const int mOff = (lane & 16) ? 8 : 0, nCol = lane & 15;
#pragma unroll
    for (int i = 0; i < 2; ++i) {
#pragma unroll
        for (int j = 0; j < 4; ++j) {
            const int p = nSub + j * 16 + nCol;                 // pixel within 128-tile
            const int y = yBase + (p >> 6), x = p & 63;
#pragma unroll
            for (int r = 0; r < 8; ++r) {
                const int oc = mBase + mSub + i * 16 + mOff + r;
                const long long idx = (((long long)b * 512 + oc) << 12) + y * 64 + x;
                float v = acc[i][j][r] + bias[oc];
                if (resid) v += resid[idx];
                outp[idx] = v;
            }
        }
    }
}

// ---------------- GroupNorm ----------------
__global__ __launch_bounds__(256) void gn_stats(const float* __restrict__ x, float* __restrict__ stats) {
    const int bg = blockIdx.x;                     // group region = 16*4096 contiguous floats
    const float* p = x + (long long)bg * 65536;
    float s = 0.f, s2 = 0.f;
    for (int i = threadIdx.x; i < 65536; i += 256) { float v = p[i]; s += v; s2 += v * v; }
    __shared__ float sh[256], sh2[256];
    sh[threadIdx.x] = s; sh2[threadIdx.x] = s2; __syncthreads();
    for (int st = 128; st > 0; st >>= 1) {
        if (threadIdx.x < st) { sh[threadIdx.x] += sh[threadIdx.x + st]; sh2[threadIdx.x] += sh2[threadIdx.x + st]; }
        __syncthreads();
    }
    if (threadIdx.x == 0) {
        float mean = sh[0] * (1.f / 65536.f);
        float var  = sh2[0] * (1.f / 65536.f) - mean * mean;
        stats[bg * 2] = mean;
        stats[bg * 2 + 1] = rsqrtf(var + 1e-5f);
    }
}

// NCHW f32 in -> NHWC f16 out, optional SiLU
__global__ __launch_bounds__(256) void gn_apply(
    const float* __restrict__ x, const float* __restrict__ stats,
    const float* __restrict__ sc, const float* __restrict__ bi,
    _Float16* __restrict__ out, int doSilu)
{
    const long long t = (long long)blockIdx.x * 256 + threadIdx.x;   // < 4*4096*512
    const int c = (int)(t & 511);
    const int pix = (int)((t >> 9) & 4095);
    const int b = (int)(t >> 21);
    const float v = x[(((long long)b * 512 + c) << 12) + pix];
    const int bg = b * 32 + (c >> 4);
    float y = (v - stats[bg * 2]) * stats[bg * 2 + 1] * sc[c] + bi[c];
    if (doSilu) y = y / (1.f + expf(-y));
    out[t] = (_Float16)y;
}

// ---------------- row softmax over 4096 f16 elements, in place ----------------
__global__ __launch_bounds__(256) void softmax_rows(_Float16* __restrict__ S) {
    _Float16* p = S + (long long)blockIdx.x * 4096;
    float v[16];
    float mx = -3.4e38f;
#pragma unroll
    for (int i = 0; i < 16; ++i) { v[i] = (float)p[threadIdx.x + i * 256]; mx = fmaxf(mx, v[i]); }
    __shared__ float sh[256];
    sh[threadIdx.x] = mx; __syncthreads();
    for (int st = 128; st > 0; st >>= 1) {
        if (threadIdx.x < st) sh[threadIdx.x] = fmaxf(sh[threadIdx.x], sh[threadIdx.x + st]);
        __syncthreads();
    }
    mx = sh[0]; __syncthreads();
    float s = 0.f;
#pragma unroll
    for (int i = 0; i < 16; ++i) { v[i] = expf(v[i] - mx); s += v[i]; }
    sh[threadIdx.x] = s; __syncthreads();
    for (int st = 128; st > 0; st >>= 1) {
        if (threadIdx.x < st) sh[threadIdx.x] += sh[threadIdx.x + st];
        __syncthreads();
    }
    const float inv = 1.f / sh[0];
#pragma unroll
    for (int i = 0; i < 16; ++i) p[threadIdx.x + i * 256] = (_Float16)(v[i] * inv);
}

// ---------------- weight prepack ----------------
// w [oc][ic][3][3] f32 -> wpk [oc][(ky*3+kx)*512 + ic] f16
__global__ __launch_bounds__(256) void pack_conv_w(const float* __restrict__ w, _Float16* __restrict__ o) {
    const int t = blockIdx.x * 256 + threadIdx.x;
    if (t >= 512 * 4608) return;
    const int oc = t / 4608, r = t % 4608;
    const int tap = r >> 9, ic = r & 511;
    const int ky = tap / 3, kx = tap % 3;
    o[t] = (_Float16)w[(((long long)oc * 512 + ic) * 3 + ky) * 3 + kx];
}

__global__ __launch_bounds__(256) void cast_f16(const float* __restrict__ w, _Float16* __restrict__ o, int n) {
    const int t = blockIdx.x * 256 + threadIdx.x;
    if (t < n) o[t] = (_Float16)w[t];
}

// ---------------- orchestration ----------------
extern "C" void kernel_launch(void* const* d_in, const int* in_sizes, int n_in,
                              void* d_out, int out_size, void* d_ws, size_t ws_size,
                              hipStream_t stream) {
    (void)in_sizes; (void)n_in; (void)out_size; (void)ws_size;
    const float* x = (const float*)d_in[0];
    auto F = [&](int i) { return (const float*)d_in[i]; };

    char* ws = (char*)d_ws;
    size_t off = 0;
    auto carve = [&](size_t bytes) -> void* {
        void* p = ws + off; off += (bytes + 255) & ~(size_t)255; return p;
    };
    _Float16* actA  = (_Float16*)carve((size_t)NB * HW * CCH * 2);   // NHWC f16
    _Float16* actB  = (_Float16*)carve((size_t)NB * HW * CCH * 2);
    _Float16* qH    = (_Float16*)carve((size_t)NB * HW * CCH * 2);
    _Float16* kH    = (_Float16*)carve((size_t)NB * HW * CCH * 2);
    _Float16* vTH   = (_Float16*)carve((size_t)NB * CCH * HW * 2);
    _Float16* attnH = (_Float16*)carve((size_t)NB * HW * CCH * 2);
    _Float16* S     = (_Float16*)carve((size_t)NB * HW * HW * 2);    // 134 MB
    float*    t0    = (float*)carve((size_t)NB * CCH * HW * 4);      // NCHW f32
    float*    t1    = (float*)carve((size_t)NB * CCH * HW * 4);
    _Float16* wpk1  = (_Float16*)carve((size_t)512 * 4608 * 2);
    _Float16* wpk2  = (_Float16*)carve((size_t)512 * 4608 * 2);
    _Float16* wpk3  = (_Float16*)carve((size_t)512 * 4608 * 2);
    _Float16* wpk4  = (_Float16*)carve((size_t)512 * 4608 * 2);
    _Float16* wq16  = (_Float16*)carve((size_t)512 * 512 * 2);
    _Float16* wk16  = (_Float16*)carve((size_t)512 * 512 * 2);
    _Float16* wv16  = (_Float16*)carve((size_t)512 * 512 * 2);
    _Float16* wo16  = (_Float16*)carve((size_t)512 * 512 * 2);
    float*    stats = (float*)carve(1024);

    const long long PC = (long long)HW * CCH;      // per-batch pixels*channels
    const long long PP = (long long)HW * HW;       // per-batch scores
    const float qkScale = 0.044194173824159216f;   // 1/sqrt(512)

    // --- weight prepack (deterministic, every call) ---
    pack_conv_w<<<9216, 256, 0, stream>>>(F(3),  wpk1);
    pack_conv_w<<<9216, 256, 0, stream>>>(F(7),  wpk2);
    pack_conv_w<<<9216, 256, 0, stream>>>(F(21), wpk3);
    pack_conv_w<<<9216, 256, 0, stream>>>(F(25), wpk4);
    cast_f16<<<1024, 256, 0, stream>>>(F(11), wq16, 262144);
    cast_f16<<<1024, 256, 0, stream>>>(F(13), wk16, 262144);
    cast_f16<<<1024, 256, 0, stream>>>(F(15), wv16, 262144);
    cast_f16<<<1024, 256, 0, stream>>>(F(17), wo16, 262144);

    // --- resnet block 1 ---
    gn_stats<<<128, 256, 0, stream>>>(x, stats);
    gn_apply<<<32768, 256, 0, stream>>>(x, stats, F(1), F(2), actA, 1);
    conv3x3_wmma<<<dim3(4, 128), 256, 0, stream>>>(actA, wpk1, F(4), nullptr, t0);
    gn_stats<<<128, 256, 0, stream>>>(t0, stats);
    gn_apply<<<32768, 256, 0, stream>>>(t0, stats, F(5), F(6), actB, 1);
    conv3x3_wmma<<<dim3(4, 128), 256, 0, stream>>>(actB, wpk2, F(8), x, t1);   // t1 = resnet1 out

    // --- attention ---
    gn_stats<<<128, 256, 0, stream>>>(t1, stats);
    gn_apply<<<32768, 256, 0, stream>>>(t1, stats, F(9), F(10), actA, 0);      // no SiLU
    // q, k : M = all 16384 pixels, N = 512 oc -> NHWC f16
    gemm_wmma<true><<<dim3(128, 4, 1), 256, 0, stream>>>(actA, wq16, 0, 0, 512,
        qH, 0, 512, F(12), 2, nullptr, 1.f);
    gemm_wmma<true><<<dim3(128, 4, 1), 256, 0, stream>>>(actA, wk16, 0, 0, 512,
        kH, 0, 512, F(14), 2, nullptr, 1.f);
    // v transposed: M = 512 oc, N = 4096 pixels/batch -> vT[c][pix]
    gemm_wmma<true><<<dim3(4, 32, 4), 256, 0, stream>>>(wv16, actA, 0, PC, 512,
        vTH, PC, HW, F(16), 1, nullptr, 1.f);
    // scores: S[b][i][j] = q.k / sqrt(C)
    gemm_wmma<true><<<dim3(32, 32, 4), 256, 0, stream>>>(qH, kH, PC, PC, 512,
        S, PP, HW, nullptr, 0, nullptr, qkScale);
    softmax_rows<<<16384, 256, 0, stream>>>(S);
    // out[i][c] = P . v  (A = P rows K-major, B = vT rows K-major)
    gemm_wmma<true><<<dim3(32, 4, 4), 256, 0, stream>>>(S, vTH, PP, PC, 4096,
        attnH, PC, 512, nullptr, 0, nullptr, 1.f);
    // o-projection + residual(t1) -> t0 (NCHW f32)
    gemm_wmma<false><<<dim3(4, 32, 4), 256, 0, stream>>>(wo16, attnH, 0, PC, 512,
        t0, PC, HW, F(18), 1, t1, 1.f);

    // --- resnet block 2 ---
    gn_stats<<<128, 256, 0, stream>>>(t0, stats);
    gn_apply<<<32768, 256, 0, stream>>>(t0, stats, F(19), F(20), actA, 1);
    conv3x3_wmma<<<dim3(4, 128), 256, 0, stream>>>(actA, wpk3, F(22), nullptr, t1);
    gn_stats<<<128, 256, 0, stream>>>(t1, stats);
    gn_apply<<<32768, 256, 0, stream>>>(t1, stats, F(23), F(24), actB, 1);
    conv3x3_wmma<<<dim3(4, 128), 256, 0, stream>>>(actB, wpk4, F(26), t0, (float*)d_out);
}
